// Step_1_1597727834229
// MI455X (gfx1250) — matrix operand order repaired and checked by reading, not compiled, not run on hardware
//
#include <hip/hip_runtime.h>
#include <hip/hip_bf16.h>
#include <stdint.h>

// ---------------------------------------------------------------------------
// Span-feature extraction (BERT span conv bank) for MI455X / gfx1250.
// Compute-bound (2.6e11 FLOPs, ~51MB bf16 working set -> L2 resident).
// bf16 WMMA w/ f32 accumulate; A tiles staged to LDS via the Tensor Data
// Mover (TDM) when the builtin is available, manual staging otherwise.
// ---------------------------------------------------------------------------

typedef __attribute__((ext_vector_type(16))) __bf16 v16bf;
typedef __attribute__((ext_vector_type(8)))  float  v8f;
typedef __attribute__((ext_vector_type(4)))  unsigned int u32x4;
typedef __attribute__((ext_vector_type(8)))  int          i32x8;
typedef __attribute__((ext_vector_type(4)))  int          i32x4;

constexpr int BB = 32;    // batch
constexpr int LL = 200;   // seq len
constexpr int DD = 768;   // hidden
constexpr int KK = 8;     // max span length
constexpr int SSn = 1556; // spans per sentence

#if defined(__AMDGCN__) && __has_builtin(__builtin_amdgcn_tensor_load_to_lds) && \
    __has_builtin(__builtin_amdgcn_s_wait_tensorcnt)
#define USE_TDM 1
#else
#define USE_TDM 0
#endif

// ---- helpers --------------------------------------------------------------

__device__ __forceinline__ v16bf load16(const __bf16* p0, const __bf16* p1) {
  v16bf r;
  reinterpret_cast<uint4*>(&r)[0] = *reinterpret_cast<const uint4*>(p0);
  reinterpret_cast<uint4*>(&r)[1] = *reinterpret_cast<const uint4*>(p1);
  return r;
}

#if USE_TDM
// 2D TDM load: tile_dim = 32 x 128 rows of bf16, row stride = DD elements.
// rows_valid clips dim1 so out-of-tensor rows are zero-filled by the TDM.
__device__ __forceinline__ void tdm_load_2d(unsigned lds_off,
                                            unsigned long long gaddr,
                                            int rows_valid) {
  u32x4 g0;
  g0[0] = 1u;                                          // count=1 (valid), user mode
  g0[1] = lds_off;                                     // lds_addr (bytes)
  g0[2] = (unsigned)(gaddr & 0xFFFFFFFFull);           // global_addr[31:0]
  g0[3] = (unsigned)((gaddr >> 32) & 0x1FFFFFFull)     // global_addr[56:32]
        | (2u << 30);                                  // type=2 ("image")
  i32x8 g1;
  g1[0] = 0x00010000;                                  // data_size=1 -> 2 bytes
  g1[1] = (int)(32u << 16);                            // tensor_dim0[15:0] = 32
  g1[2] = (int)(((unsigned)rows_valid & 0xFFFFu) << 16); // dim0 hi=0 | tensor_dim1 lo
  g1[3] = (int)((((unsigned)rows_valid >> 16) & 0xFFFFu) | (32u << 16)); // dim1 hi | tile_dim0=32
  g1[4] = 128;                                         // tile_dim1=128, tile_dim2=0
  g1[5] = DD;                                          // tensor_dim0_stride = 768
  g1[6] = 0;                                           // stride hi | dim1_stride lo
  g1[7] = 0;
  i32x4 z4 = {0, 0, 0, 0};
#if __clang_major__ >= 23
  i32x8 z8 = {0, 0, 0, 0, 0, 0, 0, 0};
  __builtin_amdgcn_tensor_load_to_lds(g0, g1, z4, z4, z8, 0);
#else
  __builtin_amdgcn_tensor_load_to_lds(g0, g1, z4, z4, 0);
#endif
}
#endif

// ---- 1) per-batch span offsets -------------------------------------------

__global__ void k_offsets(const int* __restrict__ mask, int* __restrict__ offs) {
  int b = threadIdx.x;
  if (b >= BB) return;
  int n = 0;
  for (int l = 0; l < LL; ++l) n += mask[b * LL + l];
  int acc = 0;
  offs[b * (KK + 1)] = 0;
  for (int j = 0; j < KK; ++j) {
    int c = n - j - 2;
    if (c < 0) c = 0;
    acc += c;
    offs[b * (KK + 1) + j + 1] = acc;
  }
}

// ---- 2) features f32 -> bf16 ---------------------------------------------

__global__ void k_cvt_feat(const float* __restrict__ f, __bf16* __restrict__ o, int n) {
  int i = blockIdx.x * blockDim.x + threadIdx.x;
  if (i < n) o[i] = (__bf16)f[i];
}

// ---- 3) weights f32 [O][I][k] -> bf16 WT[O][h*D + i] (N-major, K-contig) --

__global__ void k_cvt_w(const float* __restrict__ w, __bf16* __restrict__ wt,
                        int k, int total) {
  int i = blockIdx.x * blockDim.x + threadIdx.x;
  if (i >= total) return;
  int rk = DD * k;
  int o  = i / rk;
  int t  = i - o * rk;
  int h  = t / DD;
  int ii = t - h * DD;
  wt[i] = (__bf16)w[o * rk + ii * k + h];
}

// ---- 4) valid mask, zero invalid rows, copy unigram (k=1) rows ------------

__global__ void k_base(const float* __restrict__ feat, const int* __restrict__ offs,
                       float* __restrict__ out, int* __restrict__ valid) {
  int s = blockIdx.x, b = blockIdx.y;
  const int* ob = offs + b * (KK + 1);
  int total = ob[KK];
  float* orow = out + ((size_t)b * SSn + s) * DD;
  if (s >= total) {
    for (int d = threadIdx.x; d < DD; d += blockDim.x) orow[d] = 0.f;
    if (threadIdx.x == 0) valid[b * SSn + s] = 0;
    return;
  }
  if (threadIdx.x == 0) valid[b * SSn + s] = 1;
  int kidx = 0;
  for (int j = 1; j < KK; ++j) kidx += (ob[j] <= s);
  if (kidx == 0) {  // unigram: plain feature row, pos = s + 1
    const float* fr = feat + ((size_t)b * LL + (s + 1)) * DD;
    for (int d = threadIdx.x; d < DD; d += blockDim.x) orow[d] = fr[d];
  }
  // kidx >= 1 rows are fully written by the WMMA conv kernel
}

// ---- 5) conv-as-GEMM: bf16 WMMA, TDM-staged A tiles, fused scatter --------
// grid: x = 6 (N tiles of 128), y = 2 (M tiles of 128 rows), z = b*7 + c
// block: 256 threads = 8 waves (2M x 4N); wave tile = 64 rows x 32 cols
// (4 M-subtiles x 2 accumulators). A tile (128x32 bf16) double-buffered in LDS.

__global__ void __launch_bounds__(256) k_conv_wmma(
    const __bf16* __restrict__ featbf, const __bf16* __restrict__ wt0,
    const float* pb2, const float* pb3, const float* pb4, const float* pb5,
    const float* pb6, const float* pb7, const float* pb8,
    const int* __restrict__ offs, float* __restrict__ out) {
  __shared__ __bf16 tileA[2][128 * 32];  // 2 x 8KB

  const int z = blockIdx.z;
  const int b = z / 7;
  const int c = z % 7;            // conv index 0..6  (span bucket c+1)
  const int k = c + 2;            // kernel width 2..8
  const int Kred = DD * k;        // GEMM reduction length

  const __bf16* wt = wt0 + (size_t)DD * DD * ((k * (k - 1)) / 2 - 1);
  const float* bptrs[7] = {pb2, pb3, pb4, pb5, pb6, pb7, pb8};
  const float* bias = bptrs[c];

  const int* ob    = offs + b * (KK + 1);
  const int offj   = ob[c + 1];
  const int counts = ob[c + 2] - offj;

  const int tid  = threadIdx.x;
  const int wave = tid >> 5;          // wave32
  const int lane = tid & 31;
  const int half = lane >> 4;
  const int l15  = lane & 15;
  const int wm   = wave >> 2;         // 0..1 (M half of block tile)
  const int wn   = wave & 3;          // 0..3 (N group of 32)

  const int tbase = blockIdx.y * 128 + 1;        // first time-row of block tile
  const int n0    = blockIdx.x * 128 + wn * 32;  // first output channel of wave

  const __bf16* brow0 = wt + (size_t)(n0 + l15) * Kred;
  const __bf16* brow1 = wt + (size_t)(n0 + 16 + l15) * Kred;

  v8f vzero = {};
  v8f acc[4][2];
#pragma unroll
  for (int m = 0; m < 4; ++m) {
    acc[m][0] = vzero;
    acc[m][1] = vzero;
  }

#if USE_TDM
  int vr = LL - k + 1 - tbase;        // rows of tile that are in-tensor
  if (vr < 0) vr = 0;
  const unsigned long long gbase =
      (unsigned long long)(uintptr_t)(featbf + ((size_t)b * LL + tbase) * DD);
  const unsigned lb0 = (unsigned)(uintptr_t)&tileA[0][0];
  const unsigned lb1 = (unsigned)(uintptr_t)&tileA[1][0];
  if (wave == 0) tdm_load_2d(lb0, gbase, vr);   // prologue: fill buffer 0
#endif

  int buf = 0;
  for (int r0 = 0; r0 < Kred; r0 += 32) {
#if USE_TDM
    if (wave == 0) __builtin_amdgcn_s_wait_tensorcnt((short)0);
    __syncthreads();                             // tile[buf] visible to all waves
    if (wave == 0 && (r0 + 32) < Kred)           // prefetch next K-slice
      tdm_load_2d(buf ? lb0 : lb1,
                  gbase + (unsigned long long)(r0 + 32) * 2, vr);
#else
    __syncthreads();                             // prior reads of tile[buf] done
    {
      int q0 = tid * 2;                          // 512 x 16B chunks, 2 per thread
#pragma unroll
      for (int q = q0; q < q0 + 2; ++q) {
        int row = q >> 2, c16 = q & 3;
        int t = tbase + row;
        if (t > LL - k) t = LL - k;              // clamp: in-bounds, store-masked
        const __bf16* src = featbf + ((size_t)b * LL + t) * DD + r0 + c16 * 8;
        *(uint4*)(&tileA[buf][row * 32 + c16 * 8]) = *(const uint4*)src;
      }
    }
    __syncthreads();
#endif

    const __bf16* ta = &tileA[buf][0];
    // B 32x16 bf16 frags (col = l15): K[half*8+0..7], K[16+half*8+0..7] from WT rows
    v16bf b0 = load16(brow0 + r0 + half * 8, brow0 + r0 + 16 + half * 8);
    v16bf b1 = load16(brow1 + r0 + half * 8, brow1 + r0 + 16 + half * 8);
#pragma unroll
    for (int m = 0; m < 4; ++m) {
      int row = wm * 64 + m * 16 + l15;
      // A 16x32 frag: VGPR0-3 = K[half*8+0..7], VGPR4-7 = K[16+half*8+0..7]
      v16bf a = load16(ta + row * 32 + half * 8, ta + row * 32 + 16 + half * 8);
      acc[m][0] = __builtin_amdgcn_wmma_f32_16x16x32_bf16(false, a, false, b0,
                                                          (short)0, acc[m][0],
                                                          false, false);
      acc[m][1] = __builtin_amdgcn_wmma_f32_16x16x32_bf16(false, a, false, b1,
                                                          (short)0, acc[m][1],
                                                          false, false);
    }
    buf ^= 1;
  }

  const float bias0 = bias[n0 + l15];
  const float bias1 = bias[n0 + 16 + l15];

  // C/D layout: col = l15, row (within 16) = half*8 + v
#pragma unroll
  for (int m = 0; m < 4; ++m) {
#pragma unroll
    for (int v = 0; v < 8; ++v) {
      int t    = tbase + wm * 64 + m * 16 + half * 8 + v;  // conv position
      int srel = t - 1;                                    // slot within bucket
      if (srel < counts) {
        size_t s = (size_t)(offj + srel);
        float* orow = out + ((size_t)b * SSn + s) * DD;
        orow[n0 + l15]      = acc[m][0][v] + bias0;
        orow[n0 + 16 + l15] = acc[m][1][v] + bias1;
      }
    }
  }
}

// ---- launch ---------------------------------------------------------------

extern "C" void kernel_launch(void* const* d_in, const int* in_sizes, int n_in,
                              void* d_out, int out_size, void* d_ws, size_t ws_size,
                              hipStream_t stream) {
  const float* features = (const float*)d_in[0];
  const int*   mask     = (const int*)d_in[1];
  // d_in[2] = span_mask (all ones, unused)
  const float* wsrc[7];
  const float* bsrc[7];
  for (int c = 0; c < 7; ++c) {
    wsrc[c] = (const float*)d_in[3 + c];   // w2..w8
    bsrc[c] = (const float*)d_in[10 + c];  // b2..b8
  }

  // workspace layout: [offsets: 1KB][featbf: B*L*D bf16][WT: 35*D*D bf16]
  int*    offs   = (int*)d_ws;
  char*   wsb    = (char*)d_ws;
  __bf16* featbf = (__bf16*)(wsb + 1024);
  __bf16* wtb    = (__bf16*)(wsb + 1024 + (size_t)BB * LL * DD * 2);

  float* out   = (float*)d_out;
  int*   valid = (int*)d_out + (size_t)BB * SSn * DD;

  k_offsets<<<1, 32, 0, stream>>>(mask, offs);

  const int nf = BB * LL * DD;
  k_cvt_feat<<<(nf + 255) / 256, 256, 0, stream>>>(features, featbf, nf);

  for (int c = 0; c < 7; ++c) {
    int k = c + 2;
    int tot = DD * DD * k;
    __bf16* dst = wtb + (size_t)DD * DD * ((k * (k - 1)) / 2 - 1);
    k_cvt_w<<<(tot + 255) / 256, 256, 0, stream>>>(wsrc[c], dst, k, tot);
  }

  dim3 gb(SSn, BB);
  k_base<<<gb, 256, 0, stream>>>(features, offs, out, valid);

  dim3 gg(DD / 128, 2, BB * 7);
  k_conv_wmma<<<gg, 256, 0, stream>>>(featbf, wtb,
                                      bsrc[0], bsrc[1], bsrc[2], bsrc[3],
                                      bsrc[4], bsrc[5], bsrc[6],
                                      offs, out);
}
